// Sent140_GRU_55070070669435
// MI455X (gfx1250) — matrix operand order repaired and checked by reading, not compiled, not run on hardware
//
#include <hip/hip_runtime.h>
#include <hip/hip_bf16.h>
#include <stdint.h>

typedef __attribute__((ext_vector_type(16))) __bf16    v16bf;
typedef __attribute__((ext_vector_type(2)))  __bf16    bf16x2;
typedef __attribute__((ext_vector_type(8)))  float     v8f;
typedef __attribute__((ext_vector_type(2)))  float     f32x2;
typedef __attribute__((ext_vector_type(4)))  unsigned  u32x4;

#define WMMA_BF16(A,B,C) \
  __builtin_amdgcn_wmma_f32_16x16x32_bf16(false,(A),false,(B),(short)0,(C),false,false)

// Problem sizes
static constexpr int kT = 512, kB = 256, kIn = 300, kH = 128, kG = 384;

// Workspace layout (bytes)
//  XG : per-(t, btile16, wave8, gate3) 16x16 bf16 tile, lane-major (32 lanes x 16B)
static constexpr size_t XG_TILES   = (size_t)kT * 16 * 8 * 3;
static constexpr size_t XG_BYTES   = XG_TILES * 512;          // ~100.7 MB
static constexpr size_t BIH_OFF    = XG_BYTES;
static constexpr size_t BIH_BYTES  = (size_t)10 * 24 * 1024;  // 10 ksteps x 24 ntiles, 1KB/tile
static constexpr size_t BHH_OFF    = BIH_OFF + BIH_BYTES;
static constexpr size_t BHH_BYTES  = (size_t)4 * 24 * 1024;   // 4 ksteps x 24 ntiles
static constexpr size_t HOUT_OFF   = BHH_OFF + BHH_BYTES;
static constexpr size_t HOUT_BYTES = (size_t)kB * kH * 4;
static constexpr size_t WS_NEED    = HOUT_OFF + HOUT_BYTES;

// ---------- scalar helpers ----------
// pack two f32 -> one dword of bf16x2.  Prefer the native packed convert
// builtin; otherwise a vector fptrunc (RNE) which the backend can select to
// v_cvt_pk_bf16_f32 when available.
__device__ __forceinline__ unsigned pack2bf(float lo, float hi) {
#if __has_builtin(__builtin_amdgcn_cvt_pk_bf16_f32)
  auto p = __builtin_amdgcn_cvt_pk_bf16_f32(lo, hi);
  return __builtin_bit_cast(unsigned, p);
#else
  bf16x2 p = __builtin_convertvector((f32x2){lo, hi}, bf16x2);
  return __builtin_bit_cast(unsigned, p);
#endif
}
__device__ __forceinline__ float bfu_lo(unsigned u) {   // bf16 in [15:0] -> f32
  return __builtin_bit_cast(float, u << 16);
}
__device__ __forceinline__ float bfu_hi(unsigned u) {   // bf16 in [31:16] -> f32
  return __builtin_bit_cast(float, u & 0xFFFF0000u);
}
__device__ __forceinline__ float sigm_f(float x) {
  return __builtin_amdgcn_rcpf(1.f + __expf(-x));        // v_exp_f32 + v_rcp_f32
}
__device__ __forceinline__ float tanh_f(float x) {
  float e = __expf(2.f * x);
  return 1.f - 2.f * __builtin_amdgcn_rcpf(e + 1.f);     // exact at +/-inf limits
}

// ============================================================================
// Kernel 0: pre-swizzle W_ih / W_hh into per-lane WMMA B-fragment layout (bf16)
// B-frag per lane: col n = lane&15, K range = 16*(lane>=16) + [0..16) contiguous.
// Tile = 32 lanes x 16 halves = 1KB.  W_ih K padded 300 -> 320 with zeros.
// ============================================================================
__global__ void prep_kernel(const float* __restrict__ W_ih,
                            const float* __restrict__ W_hh,
                            unsigned* __restrict__ bih,
                            unsigned* __restrict__ bhh) {
  int tid  = blockIdx.x * 256 + threadIdx.x;
  int lane = tid & 31;
  int tile = tid >> 5;
  int n_lo = lane & 15;
  int k_hi = (lane >> 4) << 4;
  if (tile < 240) {                       // W_ih: 10 ksteps x 24 ntiles
    int ks = tile / 24, nt = tile % 24;
    int n = nt * 16 + n_lo;
    int kb = ks * 32 + k_hi;
    unsigned* dst = bih + (size_t)tile * 256 + lane * 8;   // dwords
    #pragma unroll
    for (int j = 0; j < 8; ++j) {
      int k0 = kb + 2 * j;
      float f0 = (k0     < kIn) ? W_ih[(size_t)n * kIn + k0]     : 0.f;
      float f1 = (k0 + 1 < kIn) ? W_ih[(size_t)n * kIn + k0 + 1] : 0.f;
      dst[j] = pack2bf(f0, f1);
    }
  } else if (tile < 336) {                // W_hh: 4 ksteps x 24 ntiles
    int t2 = tile - 240;
    int ks = t2 / 24, nt = t2 % 24;
    int n = nt * 16 + n_lo;
    int kb = ks * 32 + k_hi;
    unsigned* dst = bhh + (size_t)t2 * 256 + lane * 8;
    #pragma unroll
    for (int j = 0; j < 8; ++j)
      dst[j] = pack2bf(W_hh[(size_t)n * kH + kb + 2 * j],
                       W_hh[(size_t)n * kH + kb + 2 * j + 1]);
  }
}

// ============================================================================
// Kernel 1: xg[t][b][g] = x[b,t,:] @ W_ih[g,:]^T + b_ih[g]  (bf16 WMMA, f32 acc)
// GEMM rows ordered [t][b] so D-tile rows = batch.  WG: 64 rows x 384 cols,
// 8 waves: wave w owns ntiles {w, 8+w, 16+w} x 4 M-subtiles (12 acc tiles).
// Double-buffered LDS pipeline: 1 barrier / k-step, x prefetched into regs.
// Output stored non-temporally, directly in the scan kernel's fragment layout.
// ============================================================================
__global__ __launch_bounds__(256) void xgemm_kernel(
    const float* __restrict__ x, const float* __restrict__ b_ih,
    const unsigned short* __restrict__ bih, unsigned short* __restrict__ xg) {
  __shared__ unsigned short At[2][64 * 48]; // 64 rows x 32 K bf16, stride 48 halves (96B)

  const int wg = blockIdx.x;
  const int t  = wg >> 2;
  const int b0 = (wg & 3) * 64;
  const int tid  = threadIdx.x;
  const int lane = tid & 31;
  const int w    = tid >> 5;
  const int col  = lane & 15;
  const int hi8  = (lane >> 4) * 8;

  // accumulators pre-loaded with b_ih broadcast
  v8f acc[4][3];
  #pragma unroll
  for (int g3 = 0; g3 < 3; ++g3) {
    float bv = b_ih[(g3 * 8 + w) * 16 + col];
    #pragma unroll
    for (int m = 0; m < 4; ++m)
      #pragma unroll
      for (int j = 0; j < 8; ++j) acc[m][g3][j] = bv;
  }

  const int r  = tid >> 2;                // staging row 0..63
  const int cq = (tid & 3) * 8;           // staging col base (8 elems/thread)
  const float* xrow = x + ((size_t)(b0 + r) * kT + t) * kIn;

  // prefetch chunk 0 into registers (non-temporal: x is single-use)
  float xr8[8];
  #pragma unroll
  for (int j = 0; j < 8; ++j) {
    int c0 = cq + j;
    xr8[j] = (c0 < kIn) ? __builtin_nontemporal_load(&xrow[c0]) : 0.f;
  }

  for (int ks = 0; ks < 10; ++ks) {
    // pack prefetched chunk and commit to LDS buffer ks&1
    {
      unsigned pk[4];
      #pragma unroll
      for (int j = 0; j < 4; ++j) pk[j] = pack2bf(xr8[2 * j], xr8[2 * j + 1]);
      *reinterpret_cast<u32x4*>(&At[ks & 1][r * 48 + cq]) =
          (u32x4){pk[0], pk[1], pk[2], pk[3]};
    }
    __syncthreads();

    // issue next chunk's global loads early (lands during WMMA work)
    if (ks < 9) {
      #pragma unroll
      for (int j = 0; j < 8; ++j) {
        int c0 = (ks + 1) * 32 + cq + j;
        xr8[j] = (c0 < kIn) ? __builtin_nontemporal_load(&xrow[c0]) : 0.f;
      }
    }

    v16bf bfr[3];                         // B fragments (L2-resident, pre-swizzled)
    #pragma unroll
    for (int g3 = 0; g3 < 3; ++g3) {
      const u32x4* s = reinterpret_cast<const u32x4*>(
          bih + ((size_t)ks * 24 + (g3 * 8 + w)) * 512 + lane * 16);
      reinterpret_cast<u32x4*>(&bfr[g3])[0] = s[0];
      reinterpret_cast<u32x4*>(&bfr[g3])[1] = s[1];
    }
    const unsigned short* Ab = At[ks & 1];
    #pragma unroll
    for (int m = 0; m < 4; ++m) {
      v16bf af;                           // A fragment: 2 x ds_load_b128
      reinterpret_cast<u32x4*>(&af)[0] =
          *reinterpret_cast<const u32x4*>(&Ab[(m * 16 + col) * 48 + 2 * hi8]);
      reinterpret_cast<u32x4*>(&af)[1] =
          *reinterpret_cast<const u32x4*>(&Ab[(m * 16 + col) * 48 + 8 + 2 * hi8]);
      #pragma unroll
      for (int g3 = 0; g3 < 3; ++g3)
        acc[m][g3] = WMMA_BF16(af, bfr[g3], acc[m][g3]);
    }
  }

  // Store each lane's 8 f32 accs as 8 packed bf16 = one non-temporal b128,
  // landing exactly where the scan kernel's lane will b128-load them.
  #pragma unroll
  for (int m = 0; m < 4; ++m) {
    int bt = (b0 >> 4) + m;
    #pragma unroll
    for (int g3 = 0; g3 < 3; ++g3) {
      unsigned pk[4];
      #pragma unroll
      for (int j = 0; j < 4; ++j)
        pk[j] = pack2bf(acc[m][g3][2 * j], acc[m][g3][2 * j + 1]);
      size_t tile = (((size_t)t * 16 + bt) * 8 + w) * 3 + g3;
      __builtin_nontemporal_store((u32x4){pk[0], pk[1], pk[2], pk[3]},
          reinterpret_cast<u32x4*>(xg + tile * 256 + lane * 8));
    }
  }
}

// ============================================================================
// Kernel 2: sequential GRU scan.  16 WGs (one 16-row batch tile each), 8 waves.
// Wave w owns h columns [16w,16w+16): 3 gate tiles (r,z,n), W_hh fragments
// resident in VGPRs, h in f32 registers + double-buffered bf16 LDS mirror.
// Per step: 8 ds_b128 + 3 prefetched global_b128 + 12 v_wmma + gates + 1 barrier.
// ============================================================================
__global__ __launch_bounds__(256) void scan_kernel(
    const unsigned short* __restrict__ xg, const unsigned short* __restrict__ bhh,
    const float* __restrict__ b_hh, float* __restrict__ h_out) {
  __shared__ unsigned short Hs[2][16 * 136]; // 16x128 bf16, stride 136 (conflict-free b128)

  const int bt   = blockIdx.x;            // batch tile 0..15
  const int tid  = threadIdx.x;
  const int lane = tid & 31;
  const int w    = tid >> 5;
  const int col  = lane & 15;
  const int hi   = lane >> 4;
  const int hi8  = hi * 8;

  v16bf bf[3][4];                         // W_hh B-frags: resident for all 512 steps
  #pragma unroll
  for (int g3 = 0; g3 < 3; ++g3)
    #pragma unroll
    for (int k = 0; k < 4; ++k) {
      const u32x4* s = reinterpret_cast<const u32x4*>(
          bhh + ((size_t)k * 24 + (g3 * 8 + w)) * 512 + lane * 16);
      reinterpret_cast<u32x4*>(&bf[g3][k])[0] = s[0];
      reinterpret_cast<u32x4*>(&bf[g3][k])[1] = s[1];
    }
  // pre-broadcast b_hh: used directly as the C operand of the k=0 WMMA
  v8f biasv[3];
  #pragma unroll
  for (int g3 = 0; g3 < 3; ++g3) {
    float bv = b_hh[g3 * kH + w * 16 + col];
    #pragma unroll
    for (int j = 0; j < 8; ++j) biasv[g3][j] = bv;
  }

  v8f h;
  #pragma unroll
  for (int j = 0; j < 8; ++j) h[j] = 0.f;
  for (int i = tid; i < 16 * 136; i += 256) Hs[0][i] = 0;

  const size_t xbase = ((size_t)bt * 8 + w) * 3;   // tile index before t/gate terms
  // prefetch xg tiles for t = 0
  u32x4 xq[3];
  #pragma unroll
  for (int g3 = 0; g3 < 3; ++g3)
    xq[g3] = *reinterpret_cast<const u32x4*>(xg + (xbase + g3) * 256 + lane * 8);

  __syncthreads();

  int p = 0;
  for (int t = 0; t < kT; ++t) {
    const unsigned short* Hrd = Hs[p];
    v16bf af[4];                          // h A-fragments from LDS buffer p
    #pragma unroll
    for (int k = 0; k < 4; ++k) {
      reinterpret_cast<u32x4*>(&af[k])[0] =
          *reinterpret_cast<const u32x4*>(&Hrd[col * 136 + k * 32 + 2 * hi8]);
      reinterpret_cast<u32x4*>(&af[k])[1] =
          *reinterpret_cast<const u32x4*>(&Hrd[col * 136 + k * 32 + 8 + 2 * hi8]);
    }

    // issue next step's xg loads now; WMMA + gates below hide the latency
    int tn = (t + 1) & (kT - 1);
    u32x4 xqn[3];
    #pragma unroll
    for (int g3 = 0; g3 < 3; ++g3)
      xqn[g3] = *reinterpret_cast<const u32x4*>(
          xg + (((size_t)tn * 16) * 24 + xbase + g3) * 256 + lane * 8);

    v8f acc[3];
    acc[0] = WMMA_BF16(af[0], bf[0][0], biasv[0]);
    acc[1] = WMMA_BF16(af[0], bf[1][0], biasv[1]);
    acc[2] = WMMA_BF16(af[0], bf[2][0], biasv[2]);
    #pragma unroll
    for (int k = 1; k < 4; ++k) {
      acc[0] = WMMA_BF16(af[k], bf[0][k], acc[0]);
      acc[1] = WMMA_BF16(af[k], bf[1][k], acc[1]);
      acc[2] = WMMA_BF16(af[k], bf[2][k], acc[2]);
    }

    // unpack current xg tiles (loaded last iteration): 1 VALU op per element
    v8f xr[3];
    #pragma unroll
    for (int g3 = 0; g3 < 3; ++g3) {
      #pragma unroll
      for (int j = 0; j < 4; ++j) {
        xr[g3][2 * j]     = bfu_lo(xq[g3][j]);
        xr[g3][2 * j + 1] = bfu_hi(xq[g3][j]);
      }
    }

    #pragma unroll
    for (int j = 0; j < 8; ++j) {         // GRU gates (f32)
      float rg = sigm_f(xr[0][j] + acc[0][j]);
      float zg = sigm_f(xr[1][j] + acc[1][j]);
      float ng = tanh_f(xr[2][j] + rg * acc[2][j]);
      h[j] = ng + zg * (h[j] - ng);
    }

    // mirror h tile into the other LDS buffer (bf16)
    unsigned short* Hwr = Hs[p ^ 1];
    #pragma unroll
    for (int j = 0; j < 4; ++j) {
      unsigned pk = pack2bf(h[2 * j], h[2 * j + 1]);
      Hwr[(2 * j     + 8 * hi) * 136 + w * 16 + col] = (unsigned short)(pk & 0xFFFFu);
      Hwr[(2 * j + 1 + 8 * hi) * 136 + w * 16 + col] = (unsigned short)(pk >> 16);
    }

    __syncthreads();                      // single barrier per step (double buffer)
    #pragma unroll
    for (int g3 = 0; g3 < 3; ++g3) xq[g3] = xqn[g3];
    p ^= 1;
  }

  #pragma unroll
  for (int j = 0; j < 8; ++j)
    h_out[((size_t)bt * 16 + j + 8 * hi) * kH + w * 16 + col] = h[j];
}

// ============================================================================
// Kernel 3: relu -> [256x128]x[128x2] -> log_softmax (tiny, scalar)
// ============================================================================
__global__ void head_kernel(const float* __restrict__ h_out,
                            const float* __restrict__ W_out,
                            const float* __restrict__ b_out,
                            float* __restrict__ out) {
  int b = threadIdx.x;                    // 256 threads, 1 block
  float l0 = b_out[0], l1 = b_out[1];
  for (int k = 0; k < kH; ++k) {
    float f = h_out[(size_t)b * kH + k];
    f = f > 0.f ? f : 0.f;
    l0 += f * W_out[k];
    l1 += f * W_out[kH + k];
  }
  float m   = fmaxf(l0, l1);
  float lse = m + __logf(__expf(l0 - m) + __expf(l1 - m));
  out[b * 2 + 0] = l0 - lse;
  out[b * 2 + 1] = l1 - lse;
}

// ============================================================================
extern "C" void kernel_launch(void* const* d_in, const int* in_sizes, int n_in,
                              void* d_out, int out_size, void* d_ws, size_t ws_size,
                              hipStream_t stream) {
  const float* x     = (const float*)d_in[0];
  const float* W_ih  = (const float*)d_in[1];
  const float* W_hh  = (const float*)d_in[2];
  const float* b_ih  = (const float*)d_in[3];
  const float* b_hh  = (const float*)d_in[4];
  const float* W_out = (const float*)d_in[5];
  const float* b_out = (const float*)d_in[6];

  if (ws_size < WS_NEED) return;          // need ~101 MB of scratch

  char* ws = (char*)d_ws;
  unsigned short* XG   = (unsigned short*)(ws);
  unsigned short* BIH  = (unsigned short*)(ws + BIH_OFF);
  unsigned short* BHH  = (unsigned short*)(ws + BHH_OFF);
  float*          HOUT = (float*)(ws + HOUT_OFF);

  prep_kernel <<<42,   256, 0, stream>>>(W_ih, W_hh, (unsigned*)BIH, (unsigned*)BHH);
  xgemm_kernel<<<2048, 256, 0, stream>>>(x, b_ih, BIH, XG);
  scan_kernel <<<16,   256, 0, stream>>>(XG, BHH, b_hh, HOUT);
  head_kernel <<<1,    256, 0, stream>>>(HOUT, W_out, b_out, (float*)d_out);
}
